// AngularPenaltySMLoss_74079595921633
// MI455X (gfx1250) — matrix operand order repaired and checked by reading, not compiled, not run on hardware
//
#include <hip/hip_runtime.h>

// ---------------------------------------------------------------------------
// ArcFace (AngularPenaltySMLoss) for MI455X / gfx1250, wave32 + WMMA.
//   wf = normalize(x) @ W^T   (M=1024, C=85742, K=512, fp32 in/out)
//   loss = -mean( num - log(exp(num) + sum_{c!=y} exp(S*wf)) )
// GEMM: bf16 hi/lo split emulation of fp32 on V_WMMA_F32_16X16X32_BF16.
// Data movement: W tiles via TDM (TENSOR_LOAD_TO_LDS, TENSORcnt) with D#
// padding matching the 36-float LDS row stride; X tiles via
// GLOBAL_LOAD_ASYNC_TO_LDS_B128 (ASYNCcnt). Double-buffered LDS pipeline.
// Nontemporal wf stores keep W (175.6 MB) resident in the 192 MB L2.
// ---------------------------------------------------------------------------

typedef __attribute__((ext_vector_type(16))) __bf16       v16bf;
typedef __attribute__((ext_vector_type(8)))  float        v8f;
typedef __attribute__((ext_vector_type(4)))  int          v4i;
typedef __attribute__((ext_vector_type(8)))  int          v8i;
typedef __attribute__((ext_vector_type(4)))  unsigned int v4u;

typedef __attribute__((address_space(1))) v4i* gv4i_p;   // global v4i*
typedef __attribute__((address_space(3))) v4i* lv4i_p;   // LDS v4i*

#define TILE_M 128
#define TILE_N 128
#define TILE_K 32
#define LDSPAD 36            // 32 + 4 floats: conflict-free b128 row gathers

#define ARC_S 64.0f
#define ARC_M 0.5f
#define ARC_EPS 1e-7f

// --------- CDNA5 async global->LDS copy (ASYNCcnt-tracked) -----------------
#if defined(__AMDGCN__) && __has_builtin(__builtin_amdgcn_global_load_async_to_lds_b128)
#define ASYNC_COPY_B128(gsrc, ldst)                                           \
    __builtin_amdgcn_global_load_async_to_lds_b128((gv4i_p)(gsrc),            \
                                                   (lv4i_p)(ldst), 0, 0)
#else
#define ASYNC_COPY_B128(gsrc, ldst)                                           \
    asm volatile("global_load_async_to_lds_b128 %0, %1, off"                  \
                 :: "v"((unsigned int)(size_t)(lv4i_p)(ldst)),                \
                    "v"((unsigned long long)(size_t)(gsrc))                   \
                 : "memory")
#endif

#if defined(__AMDGCN__) && __has_builtin(__builtin_amdgcn_s_wait_asynccnt)
#define WAIT_ASYNC(n) __builtin_amdgcn_s_wait_asynccnt(n)
#else
#define WAIT_ASYNC(n) asm volatile("s_wait_asynccnt " #n ::: "memory")
#endif

// --------- CDNA5 Tensor Data Mover (TENSORcnt-tracked) ---------------------
#if defined(__AMDGCN__) && __has_builtin(__builtin_amdgcn_tensor_load_to_lds)
#define HAVE_TDM 1
#else
#define HAVE_TDM 0
#endif

#if defined(__AMDGCN__) && __has_builtin(__builtin_amdgcn_s_wait_tensorcnt)
#define WAIT_TENSOR(n) __builtin_amdgcn_s_wait_tensorcnt(n)
#else
#define WAIT_TENSOR(n) asm volatile("s_wait_tensorcnt " #n ::: "memory")
#endif

// ---------------------------------------------------------------------------
// Kernel 1: write xn = x / max(||x||,1e-12) into workspace; zero row sums.
// ---------------------------------------------------------------------------
__global__ __launch_bounds__(256) void arc_prep_kernel(
    const float* __restrict__ x, float* __restrict__ xn,
    float* __restrict__ row_sums, int N, int D)
{
    const int tid  = threadIdx.x;
    const int gtid = blockIdx.x * 256 + tid;
    if (gtid < N) row_sums[gtid] = 0.0f;

    const int wave = tid >> 5;
    const int lane = tid & 31;
    const int row  = blockIdx.x * 8 + wave;
    if (row >= N) return;

    float s = 0.0f;
    for (int i = lane; i < D; i += 32) {
        const float v = x[(size_t)row * D + i];
        s += v * v;
    }
    s += __shfl_xor(s, 1, 32);
    s += __shfl_xor(s, 2, 32);
    s += __shfl_xor(s, 4, 32);
    s += __shfl_xor(s, 8, 32);
    s += __shfl_xor(s, 16, 32);
    const float inv = 1.0f / fmaxf(sqrtf(s), 1e-12f);
    for (int i = lane; i < D; i += 32)
        xn[(size_t)row * D + i] = x[(size_t)row * D + i] * inv;
}

// ---------------------------------------------------------------------------
// bf16 hi/lo split helper
// ---------------------------------------------------------------------------
__device__ __forceinline__ void bf16_split(float v, __bf16& h, __bf16& l)
{
    h = (__bf16)v;
    l = (__bf16)(v - (float)h);
}

// ---------------------------------------------------------------------------
// One K-chunk of fragment building + 8 tiles x 3 WMMA products.
// ---------------------------------------------------------------------------
__device__ __forceinline__ void mma_chunk(
    const float (*__restrict__ Xs)[LDSPAD], const float (*__restrict__ Wsh)[LDSPAD],
    v8f acc[4][2], int wm, int wn, int lane)
{
    const int am  = lane & 15;                  // A: row within 16
    const int ak0 = (lane < 16) ? 0 : 16;       // A: 16 consecutive K
    const int bn  = lane & 15;                  // B: col within 16
    const int bk0 = (lane < 16) ? 0 : 16;       // B: 16 consecutive K

    v16bf Ahi[4], Alo[4], Bhi[2], Blo[2];
#pragma unroll
    for (int tm = 0; tm < 4; ++tm) {
        const float* rp = &Xs[wm * 64 + tm * 16 + am][0];
#pragma unroll
        for (int r = 0; r < 2; ++r) {           // K-runs: ak0.. and ak0+8..
            const float4 f0 = *(const float4*)(rp + ak0 + 8 * r);
            const float4 f1 = *(const float4*)(rp + ak0 + 8 * r + 4);
            const float tmp[8] = {f0.x, f0.y, f0.z, f0.w, f1.x, f1.y, f1.z, f1.w};
#pragma unroll
            for (int e = 0; e < 8; ++e) {
                __bf16 h, l;
                bf16_split(tmp[e], h, l);
                Ahi[tm][8 * r + e] = h;
                Alo[tm][8 * r + e] = l;
            }
        }
    }
#pragma unroll
    for (int tn = 0; tn < 2; ++tn) {
        const float* rp = &Wsh[wn * 32 + tn * 16 + bn][0];
#pragma unroll
        for (int r = 0; r < 4; ++r) {
            const float4 f = *(const float4*)(rp + bk0 + 4 * r);
            const float tmp[4] = {f.x, f.y, f.z, f.w};
#pragma unroll
            for (int e = 0; e < 4; ++e) {
                __bf16 h, l;
                bf16_split(tmp[e], h, l);
                Bhi[tn][4 * r + e] = h;
                Blo[tn][4 * r + e] = l;
            }
        }
    }

#pragma unroll
    for (int tm = 0; tm < 4; ++tm) {
#pragma unroll
        for (int tn = 0; tn < 2; ++tn) {
            acc[tm][tn] = __builtin_amdgcn_wmma_f32_16x16x32_bf16(
                false, Ahi[tm], false, Bhi[tn], (short)0, acc[tm][tn], false, false);
            acc[tm][tn] = __builtin_amdgcn_wmma_f32_16x16x32_bf16(
                false, Alo[tm], false, Bhi[tn], (short)0, acc[tm][tn], false, false);
            acc[tm][tn] = __builtin_amdgcn_wmma_f32_16x16x32_bf16(
                false, Ahi[tm], false, Blo[tn], (short)0, acc[tm][tn], false, false);
        }
    }
}

// ---------------------------------------------------------------------------
// Kernel 2: tiled WMMA GEMM with fused exp-row-sum epilogue.
// Block = 256 threads (8 waves). Block tile 128x128, K-chunk 32, 2 LDS bufs.
// W tile: one TDM descriptor per chunk (wave 0). X tile: async b128 copies.
// ---------------------------------------------------------------------------
__global__ __launch_bounds__(256) void arc_gemm_kernel(
    const float* __restrict__ xn, const float* __restrict__ W,
    float* __restrict__ wf, float* __restrict__ row_sums,
    int D, int C, int computeSums)
{
    __shared__ float Xs[2][TILE_M][LDSPAD];
    __shared__ float Wsh[2][TILE_N][LDSPAD];

    const int tid  = threadIdx.x;
    const int lane = tid & 31;
    const int wave = tid >> 5;
    const int wm   = wave >> 2;                 // 0..1 (row half)
    const int wn   = wave & 3;                  // 0..3 (col quarter)
    const int base_m = blockIdx.x * TILE_M;
    const int base_c = blockIdx.y * TILE_N;

    // loader mapping: 256 threads x 4 iters x 16B = one 128x32 f32 tile
    const int lrow = tid >> 3;                  // 0..31
    const int lcol = (tid & 7) * 4;             // 0,4,...,28

    const v8f vzero = {0.f, 0.f, 0.f, 0.f, 0.f, 0.f, 0.f, 0.f};
    v8f acc[4][2];
#pragma unroll
    for (int tm = 0; tm < 4; ++tm)
#pragma unroll
        for (int tn = 0; tn < 2; ++tn) acc[tm][tn] = vzero;

    // X tile: 4 per-thread async b128 copies per chunk
    auto issue_x = [&](int buf, int kko) {
#pragma unroll
        for (int i = 0; i < 4; ++i) {
            const int r = lrow + 32 * i;
            ASYNC_COPY_B128(xn + (size_t)(base_m + r) * D + kko + lcol,
                            &Xs[buf][r][lcol]);
        }
    };

#if HAVE_TDM
    // W tile: one 2D TDM descriptor (tile 32 x 128 f32, padded rows in LDS)
    auto issue_w = [&](int buf, int kko) {
        if (wave != 0) return;
        const unsigned long long gaddr =
            (unsigned long long)(size_t)(W + (size_t)base_c * D + kko);
        const unsigned lds   = (unsigned)(size_t)(lv4i_p)&Wsh[buf][0][0];
        const unsigned tdim0 = (unsigned)(D - kko);     // valid cols from tile start
        const unsigned tdim1 = (unsigned)(C - base_c);  // valid rows (edge: zero-fill)

        v4u g0;
        g0.x = 1u;                                      // count=1, user descriptor
        g0.y = lds;                                     // lds_addr (bytes)
        g0.z = (unsigned)(gaddr & 0xFFFFFFFFu);         // global_addr[31:0]
        g0.w = (unsigned)((gaddr >> 32) & 0x01FFFFFFu)  // global_addr[56:32]
             | (2u << 30);                              // type=2 ("image")

        v8i g1;
        g1[0] = (2 << 16)        // data_size: 4 bytes
              | (1 << 20)        // pad_enable
              | (4 << 22)        // pad_interval: every 32 DWORDs
              | (3 << 25);       // pad_amount: 4 DWORDs -> 36-float row stride
        g1[1] = (int)((tdim0 & 0xFFFFu) << 16);                    // tensor_dim0 lo
        g1[2] = (int)((tdim0 >> 16) | ((tdim1 & 0xFFFFu) << 16));  // dim0 hi | dim1 lo
        g1[3] = (int)((tdim1 >> 16) | (32u << 16));                // dim1 hi | tile_dim0=32
        g1[4] = 128;                                               // tile_dim1=128
        g1[5] = 512;                                               // tensor_dim0_stride = D
        g1[6] = 0;
        g1[7] = 0;

        const v4i gz4 = {0, 0, 0, 0};                   // groups 2/3: unused (2D)
        const v8i gz8 = {0, 0, 0, 0, 0, 0, 0, 0};
        __builtin_amdgcn_tensor_load_to_lds(g0, g1, gz4, gz4, gz8, 0);
    };

    issue_x(0, 0);
    issue_w(0, 0);
    for (int kk = 0; kk < D; kk += TILE_K) {
        const int  cur  = (kk >> 5) & 1;
        const bool more = (kk + TILE_K) < D;
        if (more) { issue_x(cur ^ 1, kk + TILE_K); issue_w(cur ^ 1, kk + TILE_K); }
        if (more) { WAIT_ASYNC(4); } else { WAIT_ASYNC(0); }
        if (wave == 0) { if (more) { WAIT_TENSOR(1); } else { WAIT_TENSOR(0); } }
        __syncthreads();                        // tiles for `cur` visible to all
        mma_chunk(Xs[cur], Wsh[cur], acc, wm, wn, lane);
        __syncthreads();                        // done reading before re-issue
    }
#else
    // fallback: W tile also via per-thread async copies
    auto issue_w_async = [&](int buf, int kko) {
#pragma unroll
        for (int i = 0; i < 4; ++i) {
            const int r = lrow + 32 * i;
            int gr = base_c + r;
            if (gr >= C) gr = C - 1;
            ASYNC_COPY_B128(W + (size_t)gr * D + kko + lcol, &Wsh[buf][r][lcol]);
        }
    };
    issue_x(0, 0);
    issue_w_async(0, 0);
    for (int kk = 0; kk < D; kk += TILE_K) {
        const int  cur  = (kk >> 5) & 1;
        const bool more = (kk + TILE_K) < D;
        if (more) { issue_x(cur ^ 1, kk + TILE_K); issue_w_async(cur ^ 1, kk + TILE_K); }
        if (more) { WAIT_ASYNC(8); } else { WAIT_ASYNC(0); }
        __syncthreads();
        mma_chunk(Xs[cur], Wsh[cur], acc, wm, wn, lane);
        __syncthreads();
    }
#endif

    // ---- epilogue: nontemporal store of wf, per-row exp sums ----
    const int half = (lane >> 4) << 3;          // C/D: lanes 16-31 hold rows +8
    const int ncol = lane & 15;

    float esum[4][8];
#pragma unroll
    for (int tm = 0; tm < 4; ++tm)
#pragma unroll
        for (int j = 0; j < 8; ++j) esum[tm][j] = 0.0f;

#pragma unroll
    for (int tn = 0; tn < 2; ++tn) {
        const int col = base_c + wn * 32 + tn * 16 + ncol;
        const bool ok = (col < C);
#pragma unroll
        for (int tm = 0; tm < 4; ++tm) {
#pragma unroll
            for (int j = 0; j < 8; ++j) {
                const float v = acc[tm][tn][j];
                const int row = base_m + wm * 64 + tm * 16 + half + j;
                if (ok) {
                    // streaming store: don't evict W from L2
                    __builtin_nontemporal_store(v, &wf[(size_t)row * (size_t)C + col]);
                    if (computeSums) esum[tm][j] += __expf(ARC_S * v);
                }
            }
        }
    }

    if (computeSums) {
#pragma unroll
        for (int tm = 0; tm < 4; ++tm) {
#pragma unroll
            for (int j = 0; j < 8; ++j) {
                float s = esum[tm][j];
                s += __shfl_xor(s, 1, 32);      // reduce within 16-lane group
                s += __shfl_xor(s, 2, 32);
                s += __shfl_xor(s, 4, 32);
                s += __shfl_xor(s, 8, 32);
                if (ncol == 0) {
                    const int row = base_m + wm * 64 + tm * 16 + half + j;
                    atomicAdd(&row_sums[row], s);
                }
            }
        }
    }
}

// ---------------------------------------------------------------------------
// Kernel 3: ArcFace loss reduction. One block of N threads.
// ---------------------------------------------------------------------------
__global__ __launch_bounds__(1024) void arc_loss_kernel(
    const float* __restrict__ wf, const float* __restrict__ row_sums,
    const int* __restrict__ labels, float* __restrict__ out_scalar,
    int N, int C)
{
    __shared__ float red[1024];
    const int n = threadIdx.x;
    float L = 0.0f;
    if (n < N) {
        const int   lab = labels[n];
        const float raw = wf[(size_t)n * (size_t)C + lab];
        const float t   = fminf(fmaxf(raw, -1.0f + ARC_EPS), 1.0f - ARC_EPS);
        const float num = ARC_S * cosf(acosf(t) + ARC_M);
        const float excl = row_sums[n] - expf(ARC_S * raw);
        L = num - logf(expf(num) + excl);
    }
    red[n] = L;
    __syncthreads();
    for (int s = 512; s > 0; s >>= 1) {
        if (n < s) red[n] += red[n + s];
        __syncthreads();
    }
    if (n == 0) out_scalar[0] = -red[0] / (float)N;
}

// ---------------------------------------------------------------------------
// Host launcher
// ---------------------------------------------------------------------------
extern "C" void kernel_launch(void* const* d_in, const int* in_sizes, int n_in,
                              void* d_out, int out_size, void* d_ws, size_t ws_size,
                              hipStream_t stream)
{
    const float* x      = (const float*)d_in[0];
    const float* W      = (const float*)d_in[1];
    const int*   labels = (const int*)d_in[2];

    const int D = 512;
    const int N = in_sizes[0] / D;          // 1024
    const int C = in_sizes[1] / D;          // 85742

    float* wf       = (float*)d_out;
    float* xn       = (float*)d_ws;         // N*D floats (2 MB)
    float* row_sums = xn + (size_t)N * D;   // N floats

    const bool want_loss = (out_size > N * C);   // positive==1 path

    arc_prep_kernel<<<N / 8, 256, 0, stream>>>(x, xn, row_sums, N, D);

    dim3 grid(N / TILE_M, (C + TILE_N - 1) / TILE_N);   // x fastest: L2 reuse of W
    arc_gemm_kernel<<<grid, 256, 0, stream>>>(xn, W, wf, row_sums,
                                              D, C, want_loss ? 1 : 0);

    if (want_loss) {
        arc_loss_kernel<<<1, 1024, 0, stream>>>(wf, row_sums, labels,
                                                wf + (size_t)N * (size_t)C, N, C);
    }
}
